// ATNLPmodel_51196010168747
// MI455X (gfx1250) — matrix operand order repaired and checked by reading, not compiled, not run on hardware
//
#include <hip/hip_runtime.h>
#include <hip/hip_bf16.h>
#include <math.h>

// ---------------------------------------------------------------------------
// Problem constants
// ---------------------------------------------------------------------------
#define B_     8
#define L1_    1024
#define C_     128
#define L2_    256
#define S_     16
#define NDB_   8192
#define NQ_    (B_ * S_)          // 128 queries
#define K_TOT  (C_ * L2_)         // 32768
#define KSTEPS (K_TOT / 32)       // 1024 bf16 WMMA k-steps total
#define KSEG_  4                  // K split across blocks
#define KSL_   (KSTEPS / KSEG_)   // 256 k-steps per block
#define BN_    64                 // DB rows per block (4 N-tiles per wave)
#define NQN_   (NQ_ * NDB_)       // one partial-sims plane
#define EPS_   1e-8f
#define INV16_ (1.0f / (16.0f + EPS_))

typedef __bf16  v16bf __attribute__((ext_vector_type(16)));
typedef float   v8f   __attribute__((ext_vector_type(8)));

union BF16x16 { v16bf v; unsigned int u[8]; uint4 q[2]; };

// f32 pair -> packed bf16 (truncation) : single v_perm_b32
__device__ __forceinline__ unsigned int bpack(float lo, float hi) {
    return __builtin_amdgcn_perm(__float_as_uint(hi), __float_as_uint(lo),
                                 0x07060302u);
}

// two channel bytes ([7:0],[15:8]) -> packed bf16 {1.0|0.0, 1.0|0.0}
__device__ __forceinline__ unsigned int onehot_pair(unsigned int bytes2, unsigned int c) {
    unsigned int w = 0u;
    if ((bytes2 & 0xFFu) == c)        w |= 0x00003F80u;
    if (((bytes2 >> 8) & 0xFFu) == c) w |= 0x3F800000u;
    return w;
}

// ---------------------------------------------------------------------------
// Kernel 0: zero the per-row sum-of-squares accumulator (atomically
// accumulated into each call, so it must be re-zeroed every call)
// ---------------------------------------------------------------------------
__global__ void __launch_bounds__(256)
init_kernel(float* __restrict__ ssq) {
    ssq[blockIdx.x * 256 + threadIdx.x] = 0.0f;
}

// ---------------------------------------------------------------------------
// Kernel 1: ch[q][l] = seq_input[b, clip(int(pos),0,L1-1)]
// ---------------------------------------------------------------------------
__global__ void __launch_bounds__(L2_)
ch_kernel(const int* __restrict__ seq, const int* __restrict__ kp,
          unsigned char* __restrict__ ch) {
    const int q = blockIdx.x;
    const int l = threadIdx.x;
    const int b = q >> 4;
    const int s = q & 15;
    const int start = kp[b * (S_ + 1) + s];
    int       end   = kp[b * (S_ + 1) + s + 1];
    if (end < start + 1) end = start + 1;
    const float frac = ((float)l + 0.5f) * (1.0f / (float)L2_);
    const float pos  = (float)start + (float)(end - start) * frac;
    int idx = (int)pos;
    if (idx < 0) idx = 0;
    if (idx > L1_ - 1) idx = L1_ - 1;
    ch[q * L2_ + l] = (unsigned char)seq[b * L1_ + idx];
}

// ---------------------------------------------------------------------------
// Kernel 2: pre-pack the one-hot A matrix into the exact per-lane WMMA
// operand layout. A is identical for every GEMM block, so build it ONCE
// (8 MB, L2-resident) instead of re-synthesizing it 512x in the GEMM.
// Thread t = ((qtile*1024 + ks)*32 + lane); writes that lane's 8 dwords.
// ---------------------------------------------------------------------------
__global__ void __launch_bounds__(256)
apack_kernel(const unsigned char* __restrict__ ch, uint4* __restrict__ apack) {
    const int t     = blockIdx.x * 256 + threadIdx.x;  // 0 .. 8*1024*32-1
    const int lane  = t & 31;
    const int ks    = (t >> 5) & (KSTEPS - 1);
    const int qtile = t >> 15;
    const int h     = lane >> 4;
    const int nl    = lane & 15;
    const int qrow  = qtile * 16 + nl;
    const unsigned int c  = (unsigned int)(ks >> 3);
    const int          l0 = (ks & 7) << 5;

    const unsigned char* arow = ch + qrow * L2_ + h * 16 + l0;
    const uint2 g1 = *(const uint2*)(arow);        // K j = h*16 + 0..7
    const uint2 g2 = *(const uint2*)(arow + 8);    // K j = h*16 + 8..15
    BF16x16 am;
    am.u[0] = onehot_pair(g1.x,       c);
    am.u[1] = onehot_pair(g1.x >> 16, c);
    am.u[2] = onehot_pair(g1.y,       c);
    am.u[3] = onehot_pair(g1.y >> 16, c);
    am.u[4] = onehot_pair(g2.x,       c);
    am.u[5] = onehot_pair(g2.x >> 16, c);
    am.u[6] = onehot_pair(g2.y,       c);
    am.u[7] = onehot_pair(g2.y >> 16, c);
    apack[(size_t)t * 2 + 0] = am.q[0];
    apack[(size_t)t * 2 + 1] = am.q[1];
}

// ---------------------------------------------------------------------------
// Kernel 3: streaming WMMA GEMM. Grid (NDB/64, KSEG) = 512 blocks x 8 waves.
// Block = 64 DB rows x all 128 queries x K-segment of 8192.
// Wave w: query tile w, 4 N-tiles sharing one pre-packed A load -> 4 WMMAs.
// The f32 database (1 GiB) is streamed exactly once; each wave additionally
// accumulates row sum-of-squares over a CONTIGUOUS 1/8 slice of the block's
// k-range (compile-time split -> no per-iteration exec masking), merged with
// global f32 atomics. Partial sims -> workspace (summed in finish kernel).
// ---------------------------------------------------------------------------

#define DO_TILE(ACC, SS, BROW, DOSS)                                         \
    {                                                                        \
        const float4* bp = (const float4*)(BROW + (size_t)ks * 32);          \
        __builtin_prefetch((const char*)bp + 4096, 0, 1);                    \
        const float4 f0 = bp[0], f1 = bp[1], f2 = bp[2], f3 = bp[3];         \
        BF16x16 bm;                                                          \
        bm.u[0] = bpack(f0.x, f0.y); bm.u[1] = bpack(f0.z, f0.w);            \
        bm.u[2] = bpack(f1.x, f1.y); bm.u[3] = bpack(f1.z, f1.w);            \
        bm.u[4] = bpack(f2.x, f2.y); bm.u[5] = bpack(f2.z, f2.w);            \
        bm.u[6] = bpack(f3.x, f3.y); bm.u[7] = bpack(f3.z, f3.w);            \
        if (DOSS) {                                                          \
            SS = fmaf(f0.x, f0.x, SS); SS = fmaf(f0.y, f0.y, SS);            \
            SS = fmaf(f0.z, f0.z, SS); SS = fmaf(f0.w, f0.w, SS);            \
            SS = fmaf(f1.x, f1.x, SS); SS = fmaf(f1.y, f1.y, SS);            \
            SS = fmaf(f1.z, f1.z, SS); SS = fmaf(f1.w, f1.w, SS);            \
            SS = fmaf(f2.x, f2.x, SS); SS = fmaf(f2.y, f2.y, SS);            \
            SS = fmaf(f2.z, f2.z, SS); SS = fmaf(f2.w, f2.w, SS);            \
            SS = fmaf(f3.x, f3.x, SS); SS = fmaf(f3.y, f3.y, SS);            \
            SS = fmaf(f3.z, f3.z, SS); SS = fmaf(f3.w, f3.w, SS);            \
        }                                                                    \
        ACC = __builtin_amdgcn_wmma_f32_16x16x32_bf16(                       \
                  false, am.v, false, bm.v, (short)0, ACC, false, false);    \
    }

#define KBODY(DOSS)                                                          \
    {                                                                        \
        const uint4 a0 = ap[0];                                              \
        const uint4 a1 = ap[1];                                              \
        ap += 64;                               /* 32 lanes * 2 uint4 */     \
        BF16x16 am; am.q[0] = a0; am.q[1] = a1;                              \
        DO_TILE(acc0, ss0, brow0, DOSS)                                      \
        DO_TILE(acc1, ss1, brow1, DOSS)                                      \
        DO_TILE(acc2, ss2, brow2, DOSS)                                      \
        DO_TILE(acc3, ss3, brow3, DOSS)                                      \
    }

__global__ void __launch_bounds__(256)
gemm_kernel(const uint4* __restrict__ apack,
            const float* __restrict__ db,
            float* __restrict__ psims,     // [KSEG][NQ][NDB] partials
            float* __restrict__ ssq) {     // [NDB] sum-of-squares (atomic)
    const int tid  = threadIdx.x;
    const int wave = tid >> 5;
    const int lane = tid & 31;
    const int h    = lane >> 4;                 // K half within a k-step
    const int nl   = lane & 15;                 // local output column
    const int n0   = blockIdx.x * BN_;
    const int kseg = blockIdx.y;
    const int q0   = wave * 16;

    // B streams: 16 consecutive f32 of a DB row per k-step (layout: lanes
    // 0-15 = K 0..15, lanes 16-31 = K 16..31 of column nl)
    const size_t kbase = (size_t)kseg * (KSL_ * 32) + h * 16;
    const float* __restrict__ brow0 = db + (size_t)(n0 +  0 + nl) * K_TOT + kbase;
    const float* __restrict__ brow1 = db + (size_t)(n0 + 16 + nl) * K_TOT + kbase;
    const float* __restrict__ brow2 = db + (size_t)(n0 + 32 + nl) * K_TOT + kbase;
    const float* __restrict__ brow3 = db + (size_t)(n0 + 48 + nl) * K_TOT + kbase;

    // Pre-packed A stream: 2 x uint4 per k-step, coalesced across the wave
    const uint4* __restrict__ ap =
        apack + ((((size_t)wave * KSTEPS + (size_t)kseg * KSL_) * 32) + lane) * 2;

    v8f acc0 = {0,0,0,0,0,0,0,0}, acc1 = acc0, acc2 = acc0, acc3 = acc0;
    float ss0 = 0.f, ss1 = 0.f, ss2 = 0.f, ss3 = 0.f;

    // Each wave owns a contiguous 1/8 slice of this block's k-range for the
    // sum-of-squares side channel; across 8 waves x 4 ksegs that covers the
    // full K exactly once.  Compile-time split -> no exec masking in the loop.
    const int lo = wave * (KSL_ / 8);
    const int hi = lo + (KSL_ / 8);
    int ks = 0;
    for (; ks < lo;   ++ks) KBODY(0)
    for (; ks < hi;   ++ks) KBODY(1)
    for (; ks < KSL_; ++ks) KBODY(0)

    // combine the two K-halves (lanes L and L^16 hold the same column)
    ss0 += __shfl_xor(ss0, 16, 32);
    ss1 += __shfl_xor(ss1, 16, 32);
    ss2 += __shfl_xor(ss2, 16, 32);
    ss3 += __shfl_xor(ss3, 16, 32);
    if (lane < 16) {
        atomicAdd(&ssq[n0 +  0 + nl], ss0);
        atomicAdd(&ssq[n0 + 16 + nl], ss1);
        atomicAdd(&ssq[n0 + 32 + nl], ss2);
        atomicAdd(&ssq[n0 + 48 + nl], ss3);
    }

    // C/D layout: VGPR r -> row q0 + 8*h + r ; column = n0 + tile*16 + nl
    float* __restrict__ pseg = psims + (size_t)kseg * NQN_;
    #pragma unroll
    for (int r = 0; r < 8; ++r) {
        const int q = q0 + h * 8 + r;
        float* orow = pseg + (size_t)q * NDB_;
        orow[n0 +  0 + nl] = acc0[r];
        orow[n0 + 16 + nl] = acc1[r];
        orow[n0 + 32 + nl] = acc2[r];
        orow[n0 + 48 + nl] = acc3[r];
    }
}

// ---------------------------------------------------------------------------
// Kernel 4 (fused): sum K-segment partials, apply 1/((||d||+eps)(16+eps)),
// write the sims row, and track the row max/argmax in the same sweep.
// One block per query.
// ---------------------------------------------------------------------------
__global__ void __launch_bounds__(256)
finish_kernel(const float* __restrict__ psims, const float* __restrict__ ssq,
              const int* __restrict__ classes,
              float* __restrict__ out, float* __restrict__ pred,
              float* __restrict__ unit) {
    __shared__ float sv[256];
    __shared__ int   si[256];
    const int q = blockIdx.x;
    const size_t roff = (size_t)q * NDB_;

    float best = -INFINITY;
    int   bi   = 0x7FFFFFFF;
    for (int n = threadIdx.x; n < NDB_; n += 256) {
        float s = psims[roff + n];
        s += psims[roff + n + 1 * NQN_];
        s += psims[roff + n + 2 * NQN_];
        s += psims[roff + n + 3 * NQN_];
        s *= INV16_ / (sqrtf(ssq[n]) + EPS_);
        out[roff + n] = s;
        if (s > best) { best = s; bi = n; }
    }
    sv[threadIdx.x] = best;
    si[threadIdx.x] = bi;
    __syncthreads();
    for (int off = 128; off > 0; off >>= 1) {
        if (threadIdx.x < off) {
            const float ov = sv[threadIdx.x + off];
            const int   oi = si[threadIdx.x + off];
            if (ov > sv[threadIdx.x] ||
                (ov == sv[threadIdx.x] && oi < si[threadIdx.x])) {
                sv[threadIdx.x] = ov;
                si[threadIdx.x] = oi;
            }
        }
        __syncthreads();
    }
    if (threadIdx.x == 0) {
        pred[q] = (float)classes[si[0]];
        unit[q] = sv[0];
    }
}

// ---------------------------------------------------------------------------
// Launcher.  Workspace layout (~24.1 MB):
//   [0,32K)        ch table (u8)
//   [32K,64K)      ssq[8192] (f32, atomically accumulated; zeroed each call)
//   [64K,64K+8M)   pre-packed one-hot A operand
//   [+8M, +24M)    K-segment partial sims (4 x 4 MB)
// ---------------------------------------------------------------------------
extern "C" void kernel_launch(void* const* d_in, const int* in_sizes, int n_in,
                              void* d_out, int out_size, void* d_ws, size_t ws_size,
                              hipStream_t stream) {
    const int*   seq     = (const int*)d_in[0];     // (8,1024)
    const int*   kp      = (const int*)d_in[1];     // (8,17)
    const float* db      = (const float*)d_in[2];   // (8192,128,256)
    const int*   classes = (const int*)d_in[3];     // (8192,)
    float*       out     = (float*)d_out;

    char* ws = (char*)d_ws;
    unsigned char* ch    = (unsigned char*)ws;                    // 32 KB
    float*         ssq   = (float*)(ws + 32768);                  // 32 KB
    uint4*         apack = (uint4*)(ws + 65536);                  // 8 MB
    float*         psims = (float*)(ws + 65536 + (size_t)8 * 1024 * 1024);

    init_kernel<<<NDB_ / 256, 256, 0, stream>>>(ssq);
    ch_kernel<<<NQ_, L2_, 0, stream>>>(seq, kp, ch);
    apack_kernel<<<(8 * KSTEPS * 32) / 256, 256, 0, stream>>>(ch, apack);

    dim3 ggrid(NDB_ / BN_, KSEG_);
    gemm_kernel<<<ggrid, 256, 0, stream>>>(apack, db, psims, ssq);

    float* pred = out + (size_t)NQ_ * NDB_;
    float* unit = pred + NQ_;
    finish_kernel<<<NQ_, 256, 0, stream>>>(psims, ssq, classes, out, pred, unit);
}
